// MyLayer_40681930228064
// MI455X (gfx1250) — matrix operand order repaired and checked by reading, not compiled, not run on hardware
//
#include <hip/hip_runtime.h>

#define TOK   2048
#define HDIM  4096
#define NEXP  8
#define TOPK  2
#define NINT  2048   // expert (and shared) intermediate width

typedef __attribute__((ext_vector_type(16))) __bf16        bf16x16;
typedef __attribute__((ext_vector_type(8)))  float         f32x8;
typedef __attribute__((ext_vector_type(4)))  float         f32x4;
typedef __attribute__((ext_vector_type(4)))  unsigned int  u32x4;
typedef __attribute__((ext_vector_type(2)))  unsigned int  u32x2;

__device__ __forceinline__ unsigned short f32_to_bf16(float f) {
  unsigned int u = __builtin_bit_cast(unsigned int, f);
  u += 0x7FFFu + ((u >> 16) & 1u);   // round-to-nearest-even (epilogue path)
  return (unsigned short)(u >> 16);
}

// Hot-path packed convert: round-to-nearest(+0x8000), pack two high halves with
// V_PERM_B32.  result.lo16 = bf16(a), result.hi16 = bf16(b).
__device__ __forceinline__ unsigned int pack2_bf16(float a, float b) {
  unsigned int ua = __builtin_bit_cast(unsigned int, a) + 0x8000u;
  unsigned int ub = __builtin_bit_cast(unsigned int, b) + 0x8000u;
  return __builtin_amdgcn_perm(ub, ua, 0x07060302u);  // {ub[31:16], ua[31:16]}
}

union FragU { u32x4 q[2]; bf16x16 v; };

// A fragment (16x32 bf16, wave32): m = lane&15; VGPR0..3 K=(lane/16)*8+2j,
// VGPR4..7 K=16+(lane/16)*8+2(j-4)  => two contiguous 16B chunks per lane.
__device__ __forceinline__ bf16x16 frag_A(const unsigned short* As, int lane, int kstep) {
  const int m  = lane & 15;
  const int h8 = (lane >> 4) << 3;
  const unsigned short* p = As + m * 72 + kstep * 32;
  FragU r;
  r.q[0] = *(const u32x4*)(p + h8);
  r.q[1] = *(const u32x4*)(p + 16 + h8);
  return r.v;
}

// B fragment (32x16 bf16): n = lane&15, elem 0..7 K=(lane/16)*8+elem, elem 8..15
// K=16+(lane/16)*8+(elem-8) => two contiguous 16B chunks along the W row (Bs stores W rows, K contiguous).
__device__ __forceinline__ bf16x16 frag_B(const unsigned short* Bs, int lane, int kstep, int nbase) {
  const int n    = lane & 15;
  const int koff = (lane >> 4) << 3;
  const unsigned short* p = Bs + (nbase + n) * 72 + kstep * 32 + koff;
  FragU r;
  r.q[0] = *(const u32x4*)(p);
  r.q[1] = *(const u32x4*)(p + 16);
  return r.v;
}

// ---------------- small utility kernels ----------------

__global__ void k_cvt_bf16(const float* __restrict__ x, unsigned int* __restrict__ y, int n4) {
  int i = blockIdx.x * blockDim.x + threadIdx.x;
  if (i >= n4) return;
  f32x4 f = ((const f32x4*)x)[i];
  u32x2 o = {pack2_bf16(f.x, f.y), pack2_bf16(f.z, f.w)};
  ((u32x2*)y)[i] = o;
}

__global__ void k_zero_f4(float* __restrict__ p, int n4) {
  int i = blockIdx.x * blockDim.x + threadIdx.x;
  if (i >= n4) return;
  f32x4 z = {0.f, 0.f, 0.f, 0.f};
  ((f32x4*)p)[i] = z;
}

__global__ void k_init_counts(int* counts) {
  if (threadIdx.x < NEXP) counts[threadIdx.x] = 0;
}

__global__ void k_gate(const int* __restrict__ idx, const float* __restrict__ w,
                       float* __restrict__ gate, int* __restrict__ counts) {
  int t = blockIdx.x * blockDim.x + threadIdx.x;
  if (t >= TOK) return;
  float g[NEXP];
#pragma unroll
  for (int e = 0; e < NEXP; ++e) g[e] = 0.f;
#pragma unroll
  for (int k = 0; k < TOPK; ++k) {
    int e = idx[t * TOPK + k];
    g[e] += w[t * TOPK + k];
  }
#pragma unroll
  for (int e = 0; e < NEXP; ++e) {
    gate[t * NEXP + e] = g[e];
    if (g[e] != 0.f) atomicAdd(&counts[e], 1);
  }
}

__global__ void k_prefix(const int* __restrict__ counts, int* __restrict__ segoff,
                         int* __restrict__ cursor) {
  if (threadIdx.x == 0 && blockIdx.x == 0) {
    int s = 0;
    for (int e = 0; e < NEXP; ++e) { segoff[e] = s; cursor[e] = s; s += counts[e]; }
    segoff[NEXP] = s;
  }
}

__global__ void k_fill(const float* __restrict__ gate, int* __restrict__ cursor,
                       int* __restrict__ rowids, float* __restrict__ gslot) {
  int t = blockIdx.x * blockDim.x + threadIdx.x;
  if (t >= TOK) return;
#pragma unroll
  for (int e = 0; e < NEXP; ++e) {
    float g = gate[t * NEXP + e];
    if (g != 0.f) {
      int s = atomicAdd(&cursor[e], 1);
      rowids[s] = t;
      gslot[s]  = g;
    }
  }
}

// ---------------- GEMM1: gu = X @ W13[e].T, h = silu(gate)*up ----------------
// 256 thr (8 waves); block tile M=64, N=64 of the 2048-wide intermediate.
// Wave (wm2, wn) owns m-subtiles {2*wm2, 2*wm2+1}; per subtile a gate acc and
// an up acc -> 8 WMMAs per wave per 64-wide K stage; wave-local SwiGLU.

__global__ __launch_bounds__(256) void k_moe_gemm1(
    const unsigned short* __restrict__ X,     // [TOK, HDIM] bf16
    const float* __restrict__ Wbase,          // [e][2*NINT, HDIM] fp32
    const int* __restrict__ segoff,           // [NEXP+1] or nullptr (shared)
    const int* __restrict__ rowids,           // compact slot->token, or nullptr
    unsigned short* __restrict__ Hout)        // [slots, NINT] bf16
{
  const int e    = blockIdx.z;
  const int seg0 = segoff ? segoff[e] : 0;
  const int rows = segoff ? (segoff[e + 1] - seg0) : TOK;
  const int m0   = blockIdx.y * 64;
  if (m0 >= rows) return;
  const float* W = Wbase + (size_t)e * (size_t)(2 * NINT) * HDIM;
  const int n0   = blockIdx.x * 64;

  __shared__ unsigned short As[64 * 72];
  __shared__ unsigned short Bs[128 * 72];

  const int tid  = threadIdx.x;
  const int lane = tid & 31;
  const int wave = tid >> 5;
  const int wm2 = wave >> 2, wn = wave & 3;

  const int  ar = tid >> 2;            // A stage: row 0..63
  const int  ac = (tid & 3) << 4;      // col 0,16,32,48 (16 bf16 = 32B)
  const bool avalid = (m0 + ar) < rows;
  int tok = 0;
  if (avalid) tok = rowids ? rowids[seg0 + m0 + ar] : (m0 + ar);
  const unsigned short* Xrow = X + (size_t)tok * HDIM;

  f32x8 accg[2], accu[2];
#pragma unroll
  for (int mi = 0; mi < 2; ++mi)
#pragma unroll
    for (int i = 0; i < 8; ++i) { accg[mi][i] = 0.f; accu[mi][i] = 0.f; }

  for (int kk = 0; kk < HDIM; kk += 64) {
    // stage A (already bf16): 16 halfs (2x16B) per thread
    u32x4 av0 = {0u, 0u, 0u, 0u}, av1 = {0u, 0u, 0u, 0u};
    if (avalid) {
      av0 = *(const u32x4*)(Xrow + kk + ac);
      av1 = *(const u32x4*)(Xrow + kk + ac + 8);
    }
    *(u32x4*)(&As[ar * 72 + ac])     = av0;
    *(u32x4*)(&As[ar * 72 + ac + 8]) = av1;

    // stage B: 128 W-rows (64 gate + 64 up) x 64 K, fp32 -> bf16 via v_perm
#pragma unroll
    for (int i = 0; i < 8; ++i) {
      int linear = (tid << 5) + (i << 2);   // 0..8191
      int brow = linear >> 6;
      int bc   = linear & 63;
      int wrow = (brow < 64) ? (n0 + brow) : (NINT + n0 + (brow - 64));
      const float* wp = W + (size_t)wrow * HDIM + kk + bc;
      f32x4 f = *(const f32x4*)wp;
      if (kk + 64 < HDIM) __builtin_prefetch(wp + 64, 0, 3);
      u32x2 pk = {pack2_bf16(f.x, f.y), pack2_bf16(f.z, f.w)};
      *(u32x2*)(&Bs[brow * 72 + bc]) = pk;
    }
    __syncthreads();
#pragma unroll
    for (int ks = 0; ks < 2; ++ks) {
      bf16x16 bg = frag_B(Bs, lane, ks, wn * 16);
      bf16x16 bu = frag_B(Bs, lane, ks, 64 + wn * 16);
#pragma unroll
      for (int mi = 0; mi < 2; ++mi) {
        bf16x16 a = frag_A(As + (wm2 * 2 + mi) * 16 * 72, lane, ks);
        accg[mi] = __builtin_amdgcn_wmma_f32_16x16x32_bf16(false, a, false, bg, (short)0, accg[mi], false, false);
        accu[mi] = __builtin_amdgcn_wmma_f32_16x16x32_bf16(false, a, false, bu, (short)0, accu[mi], false, false);
      }
    }
    __syncthreads();
  }

  // SwiGLU epilogue: C layout VGPR i -> row i (+8 for lanes 16..31), col=lane&15
  const int col = n0 + wn * 16 + (lane & 15);
#pragma unroll
  for (int mi = 0; mi < 2; ++mi) {
    const int rbase = (wm2 * 2 + mi) * 16 + ((lane >> 4) << 3);
#pragma unroll
    for (int i = 0; i < 8; ++i) {
      int r = m0 + rbase + i;
      if (r < rows) {
        float g = accg[mi][i];
        float u = accu[mi][i];
        float h = (g / (1.f + __expf(-g))) * u;
        Hout[(size_t)(seg0 + r) * NINT + col] = f32_to_bf16(h);
      }
    }
  }
}

// ---------------- GEMM2: out += gate * (h @ W2[e].T) ----------------
// Block tile M=64, N=64 over HDIM; wave owns 2 m-subtiles.

__global__ __launch_bounds__(256) void k_moe_gemm2(
    const unsigned short* __restrict__ Hin,   // [slots, NINT] bf16
    const float* __restrict__ Wbase,          // [e][HDIM, NINT] fp32
    const int* __restrict__ segoff,
    const int* __restrict__ rowids,           // nullptr => shared path (plain store)
    const float* __restrict__ gslot,
    float* __restrict__ Out)                  // [TOK, HDIM] fp32
{
  const int e    = blockIdx.z;
  const int seg0 = segoff ? segoff[e] : 0;
  const int rows = segoff ? (segoff[e + 1] - seg0) : TOK;
  const int m0   = blockIdx.y * 64;
  if (m0 >= rows) return;
  const float* W = Wbase + (size_t)e * (size_t)HDIM * NINT;
  const int n0   = blockIdx.x * 64;

  __shared__ unsigned short As[64 * 72];
  __shared__ unsigned short Bs[64 * 72];

  const int tid  = threadIdx.x;
  const int lane = tid & 31;
  const int wave = tid >> 5;
  const int wm2 = wave >> 2, wn = wave & 3;

  const int  ar = tid >> 2;
  const int  ac = (tid & 3) << 4;
  const bool avalid = (m0 + ar) < rows;
  const unsigned short* Arow = Hin + (size_t)(seg0 + m0 + ar) * NINT;

  f32x8 acc[2];
#pragma unroll
  for (int mi = 0; mi < 2; ++mi)
#pragma unroll
    for (int i = 0; i < 8; ++i) acc[mi][i] = 0.f;

  for (int kk = 0; kk < NINT; kk += 64) {
    u32x4 av0 = {0u, 0u, 0u, 0u}, av1 = {0u, 0u, 0u, 0u};
    if (avalid) {
      av0 = *(const u32x4*)(Arow + kk + ac);
      av1 = *(const u32x4*)(Arow + kk + ac + 8);
    }
    *(u32x4*)(&As[ar * 72 + ac])     = av0;
    *(u32x4*)(&As[ar * 72 + ac + 8]) = av1;

#pragma unroll
    for (int i = 0; i < 4; ++i) {
      int linear = (tid << 4) + (i << 2);   // 0..4095
      int brow = linear >> 6;
      int bc   = linear & 63;
      const float* wp = W + (size_t)(n0 + brow) * NINT + kk + bc;
      f32x4 f = *(const f32x4*)wp;
      if (kk + 64 < NINT) __builtin_prefetch(wp + 64, 0, 3);
      u32x2 pk = {pack2_bf16(f.x, f.y), pack2_bf16(f.z, f.w)};
      *(u32x2*)(&Bs[brow * 72 + bc]) = pk;
    }
    __syncthreads();
#pragma unroll
    for (int ks = 0; ks < 2; ++ks) {
      bf16x16 b = frag_B(Bs, lane, ks, wn * 16);
#pragma unroll
      for (int mi = 0; mi < 2; ++mi) {
        bf16x16 a = frag_A(As + (wm2 * 2 + mi) * 16 * 72, lane, ks);
        acc[mi] = __builtin_amdgcn_wmma_f32_16x16x32_bf16(false, a, false, b, (short)0, acc[mi], false, false);
      }
    }
    __syncthreads();
  }

  const int col = n0 + wn * 16 + (lane & 15);
#pragma unroll
  for (int mi = 0; mi < 2; ++mi) {
    const int rbase = (wm2 * 2 + mi) * 16 + ((lane >> 4) << 3);
#pragma unroll
    for (int i = 0; i < 8; ++i) {
      int r = m0 + rbase + i;
      if (r < rows) {
        if (rowids) {
          int   t = rowids[seg0 + r];
          float g = gslot[seg0 + r];
          __hip_atomic_fetch_add(&Out[(size_t)t * HDIM + col], g * acc[mi][i],
                                 __ATOMIC_RELAXED, __HIP_MEMORY_SCOPE_AGENT);
        } else {
          Out[(size_t)r * HDIM + col] = acc[mi][i];
        }
      }
    }
  }
}

// ---------------- launch ----------------

extern "C" void kernel_launch(void* const* d_in, const int* in_sizes, int n_in,
                              void* d_out, int out_size, void* d_ws, size_t ws_size,
                              hipStream_t stream) {
  (void)in_sizes; (void)n_in; (void)out_size; (void)ws_size;
  const float* hidden    = (const float*)d_in[0];
  const int*   topk_idx  = (const int*)  d_in[1];
  const float* topk_w    = (const float*)d_in[2];
  const float* w13       = (const float*)d_in[3];
  const float* w2        = (const float*)d_in[4];
  const float* w_gate_up = (const float*)d_in[5];
  const float* w_down    = (const float*)d_in[6];
  float* out_combined = (float*)d_out;
  float* out_shared   = (float*)d_out + (size_t)TOK * HDIM;

  char* ws = (char*)d_ws;
  size_t off = 0;
  auto take = [&](size_t b) { char* p = ws + off; off = (off + b + 255) & ~(size_t)255; return p; };
  unsigned short* x_bf16 = (unsigned short*)take((size_t)TOK * HDIM * 2);       // 16 MB
  unsigned short* h_exp  = (unsigned short*)take((size_t)TOK * TOPK * NINT * 2);// 16 MB (<= T*K slots)
  unsigned short* h_sh   = (unsigned short*)take((size_t)TOK * NINT * 2);       // 8 MB
  float* gate   = (float*)take((size_t)TOK * NEXP * 4);
  int*   rowids = (int*)  take((size_t)TOK * TOPK * 4);
  float* gslot  = (float*)take((size_t)TOK * TOPK * 4);
  int*   counts = (int*)  take(64);
  int*   segoff = (int*)  take(64);
  int*   cursor = (int*)  take(64);

  const int n4 = TOK * HDIM / 4;
  k_cvt_bf16<<<(n4 + 255) / 256, 256, 0, stream>>>(hidden, (unsigned int*)x_bf16, n4);
  k_zero_f4 <<<(n4 + 255) / 256, 256, 0, stream>>>(out_combined, n4);

  k_init_counts<<<1, 32, 0, stream>>>(counts);
  k_gate  <<<(TOK + 255) / 256, 256, 0, stream>>>(topk_idx, topk_w, gate, counts);
  k_prefix<<<1, 32, 0, stream>>>(counts, segoff, cursor);
  k_fill  <<<(TOK + 255) / 256, 256, 0, stream>>>(gate, cursor, rowids, gslot);

  // expert grouped GEMM1 (+SwiGLU): grid over max segment size; inactive tiles exit
  k_moe_gemm1<<<dim3(NINT / 64, TOK / 64, NEXP), 256, 0, stream>>>(
      x_bf16, w13, segoff, rowids, h_exp);
  // shared expert GEMM1 (dense)
  k_moe_gemm1<<<dim3(NINT / 64, TOK / 64, 1), 256, 0, stream>>>(
      x_bf16, w_gate_up, nullptr, nullptr, h_sh);

  // expert GEMM2 with gated scatter-add
  k_moe_gemm2<<<dim3(HDIM / 64, TOK / 64, NEXP), 256, 0, stream>>>(
      h_exp, w2, segoff, rowids, gslot, out_combined);
  // shared expert GEMM2 (plain store)
  k_moe_gemm2<<<dim3(HDIM / 64, TOK / 64, 1), 256, 0, stream>>>(
      h_sh, w_down, nullptr, nullptr, nullptr, out_shared);
}
